// KmeansCosineObjectTokens_6768868458545
// MI455X (gfx1250) — matrix operand order repaired and checked by reading, not compiled, not run on hardware
//
#include <hip/hip_runtime.h>

// Problem constants (match reference)
constexpr int Bq = 16;
constexpr int Hq = 64;
constexpr int Wq = 64;
constexpr int Cc = 512;   // channels
constexpr int Kc = 32;    // clusters
constexpr int Nn = Hq * Wq;          // 4096 points per batch
constexpr int ITERS = 20;

typedef _Float16 v8h  __attribute__((ext_vector_type(8)));
typedef _Float16 v16h __attribute__((ext_vector_type(16)));
typedef float    v8f  __attribute__((ext_vector_type(8)));

static __device__ __forceinline__ v16h cat8(v8h lo, v8h hi) {
  return __builtin_shufflevector(lo, hi, 0,1,2,3,4,5,6,7,8,9,10,11,12,13,14,15);
}

// ---------------------------------------------------------------------------
// Phase 0: L2-normalize every point; emit fp32 and fp16 copies.
// One wave (32 lanes) per point, 16 channels per lane.
__global__ __launch_bounds__(256) void k_normalize(const float* __restrict__ x,
                                                   float* __restrict__ xf,
                                                   _Float16* __restrict__ xfh) {
  const int point = blockIdx.x * 8 + (threadIdx.x >> 5);   // Bq*Nn / 8 blocks
  const int lane  = threadIdx.x & 31;
  const float* src = x + (size_t)point * Cc;
  float v[16];
  float ss = 0.f;
#pragma unroll
  for (int i = 0; i < 16; ++i) { float f = src[lane + i * 32]; v[i] = f; ss += f * f; }
#pragma unroll
  for (int off = 16; off >= 1; off >>= 1) ss += __shfl_xor(ss, off, 32);
  const float scale = 1.0f / fmaxf(sqrtf(ss), 1e-12f);
#pragma unroll
  for (int i = 0; i < 16; ++i) {
    const int ch = lane + i * 32;
    const float o = v[i] * scale;
    xf [(size_t)point * Cc + ch] = o;
    xfh[(size_t)point * Cc + ch] = (_Float16)o;
  }
}

// ---------------------------------------------------------------------------
// Gather initial centroids: c0[b,k] = xf[b, init_idx[b,k]]
__global__ __launch_bounds__(64) void k_gather(const float* __restrict__ xf,
                                               const int* __restrict__ init_idx,
                                               float* __restrict__ c,
                                               _Float16* __restrict__ chf) {
  const int bk  = blockIdx.x;           // Bq*Kc
  const int b   = bk / Kc;
  const int tid = threadIdx.x;          // 64
  const int idx = init_idx[bk];
  const float* src = xf + ((size_t)b * Nn + idx) * Cc;
#pragma unroll
  for (int i = 0; i < 8; ++i) {
    const int ch = tid + i * 64;
    const float f = src[ch];
    c  [(size_t)bk * Cc + ch] = f;
    chf[(size_t)bk * Cc + ch] = (_Float16)f;
  }
}

// ---------------------------------------------------------------------------
// Zero per-iteration accumulators.
__global__ __launch_bounds__(256) void k_zero(float* __restrict__ sums,
                                              float* __restrict__ counts) {
  const int i = blockIdx.x * 256 + threadIdx.x;
  if (i < Bq * Kc * Cc) sums[i] = 0.0f;
  if (i < Bq * Kc)      counts[i] = 0.0f;
}

// ---------------------------------------------------------------------------
// Assignment: sim = xf_f16 @ c_f16^T via v_wmma_f32_16x16x32_f16, argmax, count.
// Block = 256 threads = 8 waves; each wave owns TWO 16-point M-tiles (32 points)
// against all 32 clusters -> 4 independent accumulator chains per wave, and each
// B fragment (ds_load) feeds two WMMAs. Workgroup covers 256 points.
// blockIdx.x: point tile (256 points), blockIdx.y: batch.
__global__ __launch_bounds__(256) void k_assign(const _Float16* __restrict__ xfh,
                                                const _Float16* __restrict__ chf,
                                                int* __restrict__ assign,
                                                float* __restrict__ counts) {
  __shared__ _Float16 lds_c[Kc * Cc];     // 32 KB centroid stage
  __shared__ float    lds_sim[256 * Kc];  // 32 KB similarity tile

  const int b     = blockIdx.y;
  const int pbase = blockIdx.x * 256;
  const int tid   = threadIdx.x;

  // Stage this batch's f16 centroids into LDS (cluster-major [K][C]).
  {
    const uint4* src = (const uint4*)(chf + (size_t)b * Kc * Cc);
    uint4* dst = (uint4*)lds_c;
    for (int i = tid; i < (Kc * Cc) / 8; i += 256) dst[i] = src[i];
  }
  __syncthreads();

  const int lane = tid & 31;
  const int wave = tid >> 5;
  const int lh   = lane >> 4;     // half-wave selector
  const int l15  = lane & 15;

  // A fragments (16x32 f16): row = l15, halves at k + lh*8 and k + 16 + lh*8.
  const int mrow0 = pbase + wave * 32 + l15;        // M-tile 0
  const _Float16* a0p = xfh + ((size_t)b * Nn + mrow0) * Cc + lh * 8;
  const _Float16* a1p = a0p + (size_t)16 * Cc;      // M-tile 1 (rows +16)
  // B fragments (32x16 f16): column = l15, halves at k + lh*8 and k + 16 + lh*8.
  const _Float16* b0p = lds_c + (size_t)l15 * Cc + lh * 8;     // clusters 0..15
  const _Float16* b1p = b0p + (size_t)16 * Cc;                 // clusters 16..31

  v8f acc00 = {};   // M-tile0 x clusters 0..15
  v8f acc01 = {};   // M-tile0 x clusters 16..31
  v8f acc10 = {};   // M-tile1 x clusters 0..15
  v8f acc11 = {};   // M-tile1 x clusters 16..31
#pragma unroll 4
  for (int kk = 0; kk < Cc; kk += 32) {
    v16h A0 = cat8(*(const v8h*)(a0p + kk), *(const v8h*)(a0p + kk + 16));
    v16h A1 = cat8(*(const v8h*)(a1p + kk), *(const v8h*)(a1p + kk + 16));
    v16h B0 = cat8(*(const v8h*)(b0p + kk), *(const v8h*)(b0p + kk + 16));
    v16h B1 = cat8(*(const v8h*)(b1p + kk), *(const v8h*)(b1p + kk + 16));
    acc00 = __builtin_amdgcn_wmma_f32_16x16x32_f16(false, A0, false, B0,
                                                   (short)0, acc00, false, false);
    acc01 = __builtin_amdgcn_wmma_f32_16x16x32_f16(false, A0, false, B1,
                                                   (short)0, acc01, false, false);
    acc10 = __builtin_amdgcn_wmma_f32_16x16x32_f16(false, A1, false, B0,
                                                   (short)0, acc10, false, false);
    acc11 = __builtin_amdgcn_wmma_f32_16x16x32_f16(false, A1, false, B1,
                                                   (short)0, acc11, false, false);
  }

  // C/D layout: lane l15 = column N, VGPR r = row (r + lh*8).
#pragma unroll
  for (int r = 0; r < 8; ++r) {
    const int m0 = wave * 32 + lh * 8 + r;
    const int m1 = m0 + 16;
    lds_sim[m0 * Kc + l15]      = acc00[r];
    lds_sim[m0 * Kc + 16 + l15] = acc01[r];
    lds_sim[m1 * Kc + l15]      = acc10[r];
    lds_sim[m1 * Kc + 16 + l15] = acc11[r];
  }
  __syncthreads();

  // Per-point argmax (first max wins, matching jnp.argmax); all 256 threads busy.
  {
    const float* s = lds_sim + tid * Kc;
    float best = s[0];
    int bi = 0;
#pragma unroll
    for (int k = 1; k < Kc; ++k) {
      const float v = s[k];
      if (v > best) { best = v; bi = k; }
    }
    assign[(size_t)b * Nn + pbase + tid] = bi;
    atomicAdd(&counts[b * Kc + bi], 1.0f);
  }
}

// ---------------------------------------------------------------------------
// Scatter-sum: accumulate xf rows into per-cluster sums.
// LDS-private 32x512 fp32 table (64 KB); thread t exclusively owns channels
// {t, t+256} of every cluster row -> race-free RMW, no LDS atomics.
// blockIdx.x: 512-point chunk (8 per batch), blockIdx.y: batch.
__global__ __launch_bounds__(256) void k_scatter(const float* __restrict__ xf,
                                                 const int* __restrict__ assign,
                                                 float* __restrict__ sums) {
  __shared__ float lsum[Kc * Cc];   // 64 KB
  const int b     = blockIdx.y;
  const int pbase = blockIdx.x * 512;
  const int tid   = threadIdx.x;

  for (int i = tid; i < Kc * Cc; i += 256) lsum[i] = 0.0f;
  __syncthreads();

  const float* xp = xf + ((size_t)b * Nn + pbase) * Cc;
  const int*   ap = assign + (size_t)b * Nn + pbase;
  for (int p = 0; p < 512; ++p) {
    if (p + 4 < 512) {
      __builtin_prefetch(xp + (size_t)(p + 4) * Cc + tid, 0, 1);
    }
    const int a = ap[p];
    const float v0 = xp[(size_t)p * Cc + tid];
    const float v1 = xp[(size_t)p * Cc + tid + 256];
    lsum[a * Cc + tid]       += v0;
    lsum[a * Cc + tid + 256] += v1;
  }
  __syncthreads();

  float* gs = sums + (size_t)b * Kc * Cc;
  for (int i = tid; i < Kc * Cc; i += 256) atomicAdd(&gs[i], lsum[i]);
}

// ---------------------------------------------------------------------------
// Centroid update: mean (or keep old when empty), renormalize, emit f32 + f16.
// One wave per (b,k); in-place on c is safe (each channel owned by one lane,
// reads complete before writes within the lane).
__global__ __launch_bounds__(32) void k_update(const float* __restrict__ sums,
                                               const float* __restrict__ counts,
                                               float* __restrict__ c,
                                               _Float16* __restrict__ chf) {
  const int bk   = blockIdx.x;      // Bq*Kc
  const int lane = threadIdx.x;     // 32
  const float cnt = counts[bk];
  const float* s = sums + (size_t)bk * Cc;
  float v[16];
  float ss = 0.f;
#pragma unroll
  for (int i = 0; i < 16; ++i) {
    const int ch = lane + i * 32;
    const float m = (cnt > 0.0f) ? (s[ch] / cnt) : c[(size_t)bk * Cc + ch];
    v[i] = m;
    ss += m * m;
  }
#pragma unroll
  for (int off = 16; off >= 1; off >>= 1) ss += __shfl_xor(ss, off, 32);
  const float scale = 1.0f / fmaxf(sqrtf(ss), 1e-12f);
#pragma unroll
  for (int i = 0; i < 16; ++i) {
    const int ch = lane + i * 32;
    const float o = v[i] * scale;
    c  [(size_t)bk * Cc + ch] = o;
    chf[(size_t)bk * Cc + ch] = (_Float16)o;
  }
}

// ---------------------------------------------------------------------------
extern "C" void kernel_launch(void* const* d_in, const int* in_sizes, int n_in,
                              void* d_out, int out_size, void* d_ws, size_t ws_size,
                              hipStream_t stream) {
  const float* x        = (const float*)d_in[0];   // [B,H,W,C]
  const int*   init_idx = (const int*)d_in[1];     // [B,K]

  // Workspace layout
  float*    xf     = (float*)d_ws;                              // B*N*C f32 (128 MB)
  _Float16* xfh    = (_Float16*)(xf + (size_t)Bq * Nn * Cc);    // B*N*C f16 (64 MB)
  float*    c      = (float*)(xfh + (size_t)Bq * Nn * Cc);      // B*K*C f32
  _Float16* chf    = (_Float16*)(c + (size_t)Bq * Kc * Cc);     // B*K*C f16
  float*    sums   = (float*)(chf + (size_t)Bq * Kc * Cc);      // B*K*C f32
  float*    counts = sums + (size_t)Bq * Kc * Cc;               // B*K
  int*      assign = (int*)(counts + (size_t)Bq * Kc);          // B*N

  // Phase 0: normalize points.
  k_normalize<<<(Bq * Nn) / 8, 256, 0, stream>>>(x, xf, xfh);
  // Initial centroids.
  k_gather<<<Bq * Kc, 64, 0, stream>>>(xf, init_idx, c, chf);

  const dim3 gridAssign(Nn / 256, Bq);
  const dim3 gridScatter(Nn / 512, Bq);
  for (int it = 0; it < ITERS; ++it) {
    k_zero<<<(Bq * Kc * Cc + 255) / 256, 256, 0, stream>>>(sums, counts);
    k_assign<<<gridAssign, 256, 0, stream>>>(xfh, chf, assign, counts);
    k_scatter<<<gridScatter, 256, 0, stream>>>(xf, assign, sums);
    k_update<<<Bq * Kc, 32, 0, stream>>>(sums, counts, c, chf);
  }

  hipMemcpyAsync(d_out, c, (size_t)Bq * Kc * Cc * sizeof(float),
                 hipMemcpyDeviceToDevice, stream);
}
